// WeatherModel_34943853920894
// MI455X (gfx1250) — matrix operand order repaired and checked by reading, not compiled, not run on hardware
//
#include <hip/hip_runtime.h>
#include <hip/hip_fp16.h>
#include <math.h>

typedef __attribute__((ext_vector_type(16))) _Float16 v16h;
typedef __attribute__((ext_vector_type(8)))  float    v8f;
typedef __attribute__((ext_vector_type(4)))  float    v4f;

#define LSEQ 2048
#define NB   4096
#define IN_  8
#define HID  10
#define NG   40      // 4*HID gates
#define GP   48      // padded gate columns: 3 WMMA N-tiles
#define WPB  2       // waves per block
#define RPW  16      // batch rows per wave (WMMA M)

// Raw v_rcp_f32 (~1 ulp) instead of the IEEE div_scale/div_fmas/div_fixup
// sequence -- the gate GEMM is f16 anyway, full-precision division is wasted
// VALU on the sequential critical path.
__device__ __forceinline__ float fast_rcp(float v) {
  return __builtin_amdgcn_rcpf(v);
}
__device__ __forceinline__ float fast_sigmoid(float v) {
  return fast_rcp(1.0f + __expf(-v));          // mul, v_exp, add, v_rcp
}
__device__ __forceinline__ float fast_tanh(float v) {
  // tanh(v) = 1 - 2/(exp(2v)+1)
  return 1.0f - 2.0f * fast_rcp(1.0f + __expf(2.0f * v));
}

__global__ __launch_bounds__(WPB * 32) void lstm_wmma(
    const float* __restrict__ x,   const float* __restrict__ h0, const float* __restrict__ c0,
    const float* __restrict__ Wih, const float* __restrict__ Whh,
    const float* __restrict__ bih, const float* __restrict__ bhh,
    const float* __restrict__ Wfc, const float* __restrict__ bfc,
    float* __restrict__ out)
{
  // Per-wave private LDS slabs (no cross-wave sharing -> no s_barrier needed;
  // same-wave DS ops are executed in order on CDNA5).
  __shared__ float s_g[WPB][RPW][GP];   // gate tile spill [row][gate]
  __shared__ float s_h[WPB][RPW][HID];  // hidden state

  const int tid  = threadIdx.x;
  const int wave = tid >> 5;
  const int lane = tid & 31;
  const int m    = lane & 15;   // row (A) or column (B/C) index
  const int hi   = lane >> 4;   // which K half-stripe this lane carries
  const int n0   = (blockIdx.x * WPB + wave) * RPW;

  // ---- Build loop-invariant B tiles: Bt holds 32x16 f16, column g = 16t+m ----
  // element e of the v16h is K = 8*hi + e (e<8), 16 + 8*hi + (e-8) (e>=8)  (ISA 16-bit B layout)
  // K<8 : W_ih[g][K] ; 8<=K<18 : W_hh[g][K-8] ; K==18 : b_ih[g]+b_hh[g] ; else 0
  v16h B0, B1, B2;
  {
    v16h bt[3];
    for (int t = 0; t < 3; ++t) {
      const int g = t * 16 + m;
      const bool ok = (g < NG);
      v16h b;
      #pragma unroll
      for (int e = 0; e < 16; ++e) {
        const int K = (e < 8) ? (hi * 8 + e) : (16 + hi * 8 + (e - 8));
        float w = 0.0f;
        if (ok) {
          if (K < IN_)             w = Wih[g * IN_ + K];
          else if (K < IN_ + HID)  w = Whh[g * HID + (K - IN_)];
          else if (K == IN_ + HID) w = bih[g] + bhh[g];
        }
        b[e] = (_Float16)w;
      }
      bt[t] = b;
    }
    B0 = bt[0]; B1 = bt[1]; B2 = bt[2];
  }

  // ---- Each lane permanently owns 5 (row, j) LSTM cells; c stays in VGPRs ----
  int er[5], ej[5];
  float creg[5];
  #pragma unroll
  for (int q = 0; q < 5; ++q) {
    const int idx = lane * 5 + q;            // 0..159 covers 16 rows x 10 hidden
    er[q] = idx / HID;
    ej[q] = idx - er[q] * HID;
    creg[q] = c0[(size_t)(n0 + er[q]) * HID + ej[q]];
    s_h[wave][er[q]][ej[q]] = h0[(size_t)(n0 + er[q]) * HID + ej[q]];
  }
  __builtin_amdgcn_wave_barrier();

  const size_t xstride = (size_t)NB * IN_;
  const float* xp = x + (size_t)(n0 + m) * IN_;  // this lane's batch row in x

  // =================== sequential scan over L timesteps ===================
  for (int l = 0; l < LSEQ; ++l) {
    // Stream x_t once (lanes hi==0 carry the x K-block). Non-temporal: x is
    // touched exactly once, keep it out of the caches.
    v4f xa = {}, xb = {};
    if (hi == 0) {
      const v4f* p = (const v4f*)(xp + (size_t)l * xstride);
      xa = __builtin_nontemporal_load(p);
      xb = __builtin_nontemporal_load(p + 1);
      if (l + 1 < LSEQ)
        __builtin_prefetch(xp + (size_t)(l + 1) * xstride, 0, 1);
    }

    // A tile (16x32 f16): row m = batch, K = [ x(0..7) | h(8..17) | 1 | 0... ]
    // lanes 0-15: e=0..7 -> K=0..7 (x), e=8..15 -> K=16..23 (h8,h9,bias=1,0..)
    // lanes 16-31: e=0..7 -> K=8..15 (h0..h7), e=8..15 -> K=24..31 (zeros)
    v16h A;
    if (hi == 0) {
      A[0] = (_Float16)xa[0]; A[1] = (_Float16)xa[1];
      A[2] = (_Float16)xa[2]; A[3] = (_Float16)xa[3];
      A[4] = (_Float16)xb[0]; A[5] = (_Float16)xb[1];
      A[6] = (_Float16)xb[2]; A[7] = (_Float16)xb[3];
      A[8]  = (_Float16)s_h[wave][m][8];     // K=16
      A[9]  = (_Float16)s_h[wave][m][9];     // K=17
      A[10] = (_Float16)1.0f;                // K=18: bias row activation
      A[11] = (_Float16)0.0f; A[12] = (_Float16)0.0f; A[13] = (_Float16)0.0f;
      A[14] = (_Float16)0.0f; A[15] = (_Float16)0.0f;
    } else {
      #pragma unroll
      for (int e = 0; e < 8; ++e) A[e] = (_Float16)s_h[wave][m][e];  // K=8..15
      #pragma unroll
      for (int e = 8; e < 16; ++e) A[e] = (_Float16)0.0f;            // K=24..31
    }

    // gates[16x48] = A(16x32) @ [B0|B1|B2](32x48), f32 accumulate.
    v8f C0 = {}, C1 = {}, C2 = {};
    C0 = __builtin_amdgcn_wmma_f32_16x16x32_f16(false, A, false, B0, (short)0, C0, false, false);
    C1 = __builtin_amdgcn_wmma_f32_16x16x32_f16(false, A, false, B1, (short)0, C1, false, false);
    C2 = __builtin_amdgcn_wmma_f32_16x16x32_f16(false, A, false, B2, (short)0, C2, false, false);

    // Spill C tiles to LDS [row][gate]: C layout is row = v + 8*hi, col = m.
    #pragma unroll
    for (int v = 0; v < 8; ++v) {
      const int r = v + 8 * hi;
      s_g[wave][r][m]      = C0[v];
      s_g[wave][r][16 + m] = C1[v];
      s_g[wave][r][32 + m] = C2[v];
    }
    __builtin_amdgcn_wave_barrier();

    // Element-wise LSTM cell on this lane's 5 owned cells (gate order i,f,g,o).
    #pragma unroll
    for (int q = 0; q < 5; ++q) {
      const int r = er[q], j = ej[q];
      const float iv = fast_sigmoid(s_g[wave][r][j]);
      const float fv = fast_sigmoid(s_g[wave][r][HID + j]);
      const float gv = fast_tanh   (s_g[wave][r][2 * HID + j]);
      const float ov = fast_sigmoid(s_g[wave][r][3 * HID + j]);
      const float cv = fv * creg[q] + iv * gv;
      creg[q] = cv;
      s_h[wave][r][j] = ov * fast_tanh(cv);
    }
    __builtin_amdgcn_wave_barrier();
  }

  // ================= epilogue: y = h @ Wfc^T + bfc ; emit (y, h, c) =========
  float* yout = out;
  float* hout = out + (size_t)NB * IN_;
  float* cout = out + (size_t)NB * (IN_ + HID);

  #pragma unroll
  for (int q = 0; q < 4; ++q) {          // 16 rows x 8 outputs = 32 lanes x 4
    const int idx = lane * 4 + q;
    const int r = idx >> 3, k = idx & 7;
    float acc = bfc[k];
    #pragma unroll
    for (int j = 0; j < HID; ++j) acc += s_h[wave][r][j] * Wfc[k * HID + j];
    yout[(size_t)(n0 + r) * IN_ + k] = acc;
  }
  #pragma unroll
  for (int q = 0; q < 5; ++q) {
    const int r = er[q], j = ej[q];
    hout[(size_t)(n0 + r) * HID + j] = s_h[wave][r][j];
    cout[(size_t)(n0 + r) * HID + j] = creg[q];
  }
}

extern "C" void kernel_launch(void* const* d_in, const int* in_sizes, int n_in,
                              void* d_out, int out_size, void* d_ws, size_t ws_size,
                              hipStream_t stream) {
  const float* x   = (const float*)d_in[0];
  const float* h0  = (const float*)d_in[1];
  const float* c0  = (const float*)d_in[2];
  const float* Wih = (const float*)d_in[3];
  const float* Whh = (const float*)d_in[4];
  const float* bih = (const float*)d_in[5];
  const float* bhh = (const float*)d_in[6];
  const float* Wfc = (const float*)d_in[7];
  const float* bfc = (const float*)d_in[8];
  float* out = (float*)d_out;

  dim3 grid(NB / (WPB * RPW));   // 128 blocks
  dim3 block(WPB * 32);          // 64 threads = 2 waves
  hipLaunchKernelGGL(lstm_wmma, grid, block, 0, stream,
                     x, h0, c0, Wih, Whh, bih, bhh, Wfc, bfc, out);
}
